// HymbaOfficialAttention_49950469653320
// MI455X (gfx1250) — matrix operand order repaired and checked by reading, not compiled, not run on hardware
//
#include <hip/hip_runtime.h>
#include <hip/hip_bf16.h>

typedef _Float16 half_t;
typedef __attribute__((ext_vector_type(16))) _Float16 v16h;
typedef __attribute__((ext_vector_type(8)))  _Float16 v8h;
typedef __attribute__((ext_vector_type(8)))  float    v8f;

#define T_LEN     4096
#define NUM_HEADS 16
#define NUM_KVH   4
#define HEAD_DIM  128
#define WINDOW    1024
#define NUM_META  128
// 1/sqrt(128) * log2(e): softmax computed in base 2 so v_exp_f32 is used bare.
#define QSCALE    0.12753139905322704f

static __device__ inline v8f zero_v8f() {
  v8f z;
#pragma unroll
  for (int i = 0; i < 8; ++i) z[i] = 0.0f;
  return z;
}

static __device__ inline v16h cat16(v8h a, v8h b) {
  return __builtin_shufflevector(a, b, 0,1,2,3,4,5,6,7,8,9,10,11,12,13,14,15);
}

// 2^x lowered directly to v_exp_f32 (the HW transcendental is base-2).
static __device__ inline float fast_exp2(float x) {
#if __has_builtin(__builtin_amdgcn_exp2f)
  return __builtin_amdgcn_exp2f(x);
#else
  return exp2f(x);
#endif
}

// Max-reduce across each 16-lane half of the wave. v_permlane16_b32 keeps the
// DS pipe free for the P transpose (falls back to ds_bpermute shuffles).
static __device__ inline float lane16_max(float x) {
#if __has_builtin(__builtin_amdgcn_permlane16)
#define PL16(v, lo, hi)                                                        \
  __int_as_float(__builtin_amdgcn_permlane16(__float_as_int(v),                \
                                             __float_as_int(v), (int)(lo),    \
                                             (int)(hi), false, false))
  x = fmaxf(x, PL16(x, 0x67452301u, 0xEFCDAB89u));  // xor 1
  x = fmaxf(x, PL16(x, 0x54761032u, 0xDCFE98BAu));  // xor 2
  x = fmaxf(x, PL16(x, 0x32107654u, 0xBA98FEDCu));  // xor 4
  x = fmaxf(x, PL16(x, 0xFEDCBA98u, 0x76543210u));  // xor 8
#undef PL16
#else
#pragma unroll
  for (int off = 1; off < 16; off <<= 1)
    x = fmaxf(x, __shfl_xor(x, off, 16));
#endif
  return x;
}

// ---------------------------------------------------------------------------
// Kernel 1: RMSNorm + RoPE for Q/K (fp32 math, f16 out), V -> f16 transposed.
// Q is pre-multiplied by log2(e)/sqrt(HEAD_DIM) (base-2 softmax).
// grid = (T, 24): y in [0,16) Q heads, [16,20) K heads, [20,24) V heads.
// ---------------------------------------------------------------------------
__global__ void hymba_prep_kernel(const float* __restrict__ q,
                                  const float* __restrict__ k,
                                  const float* __restrict__ v,
                                  const float* __restrict__ qw,
                                  const float* __restrict__ kw,
                                  const int*   __restrict__ pos,
                                  half_t* __restrict__ qh,
                                  half_t* __restrict__ kh,
                                  half_t* __restrict__ vT) {
  const int t  = blockIdx.x;
  const int hk = blockIdx.y;
  const int d  = threadIdx.x;  // 0..127

  if (hk >= 20) {  // V: convert + transpose to [kvh][128][T]
    const int kvh = hk - 20;
    float x = v[(size_t)t * (NUM_KVH * HEAD_DIM) + kvh * HEAD_DIM + d];
    vT[((size_t)kvh * HEAD_DIM + d) * T_LEN + t] = (half_t)x;
    return;
  }

  __shared__ float red[128];
  __shared__ float xsh[128];

  const bool isQ = (hk < 16);
  const float* src = isQ ? (q + (size_t)t * (NUM_HEADS * HEAD_DIM) + hk * HEAD_DIM)
                         : (k + (size_t)t * (NUM_KVH * HEAD_DIM) + (hk - 16) * HEAD_DIM);
  const float* w = isQ ? qw : kw;

  float x = src[d];
  red[d] = x * x;
  __syncthreads();
#pragma unroll
  for (int s = 64; s > 0; s >>= 1) {
    if (d < s) red[d] += red[d + s];
    __syncthreads();
  }
  const float var = red[0] * (1.0f / 128.0f);
  const float xn  = w[d] * x * rsqrtf(var + 1e-6f);
  xsh[d] = xn;
  __syncthreads();

  const int   p  = pos[t];
  const int   fi = d & 63;
  const float inv_freq = __powf(10000.0f, -(float)(2 * fi) * (1.0f / 128.0f));
  const float ang = (float)p * inv_freq;
  float sn, cs;
  __sincosf(ang, &sn, &cs);
  const float other = (d < 64) ? -xsh[d + 64] : xsh[d - 64];
  float outv = xn * cs + other * sn;

  if (isQ) {
    outv *= QSCALE;  // fold softmax scale + log2(e) into Q
    qh[((size_t)hk * T_LEN + t) * HEAD_DIM + d] = (half_t)outv;
  } else {
    kh[((size_t)(hk - 16) * T_LEN + t) * HEAD_DIM + d] = (half_t)outv;
  }
}

// ---------------------------------------------------------------------------
// Kernel 2: flash attention, one wave32 per (16-query tile, head).
// WMMA 16x16x32 f16->f32 for Q*K^T, P*V, and the P-rowsum (P @ ones).
// Softmax in base 2 (scale folded into Q), masked scores are selected to
// -1e30 (cndmask, no add).
// ---------------------------------------------------------------------------
__global__ __launch_bounds__(32)
void hymba_attn_kernel(const half_t* __restrict__ qh,
                       const half_t* __restrict__ kh,
                       const half_t* __restrict__ vT,
                       float* __restrict__ out) {
  const int lane  = threadIdx.x;          // 0..31
  const int q0    = blockIdx.x * 16;
  const int h     = blockIdx.y;
  const int kvh   = h >> 2;               // GQA: repeat-interleave -> h/4
  const int n     = lane & 15;
  const int hi    = (lane >= 16) ? 1 : 0;
  const int base8 = hi ? 8 : 0;           // unused (A fragments also use koff)
  const int koff  = hi ? 16 : 0;          // B-fragment K sub-offset
  const int mbase = hi ? 8 : 0;           // C-fragment row base

  __shared__ half_t plds[16 * 32];

  // ---- Q A-fragments: 4 chunks of K=32 over head dim ----
  const half_t* qrow = qh + ((size_t)h * T_LEN + (q0 + n)) * HEAD_DIM;
  v16h qa[4];
#pragma unroll
  for (int c = 0; c < 4; ++c) {
    v8h s0 = *(const v8h*)(qrow + 32 * c + koff);
    v8h s1 = *(const v8h*)(qrow + 32 * c + koff + 8);
    qa[c] = cat16(s0, s1);
  }

  // all-ones B fragment (layout independent) for the rowsum WMMA
  v16h onesf;
#pragma unroll
  for (int i = 0; i < 16; ++i) onesf[i] = (half_t)1.0f;

  v8f acc[8];
  v8f accl = zero_v8f();  // running softmax denominator, one value per row
  float m_run[8];
#pragma unroll
  for (int i = 0; i < 8; ++i) {
    acc[i] = zero_v8f();
    m_run[i] = -3.0e38f;
  }

  const half_t* kbase = kh + (size_t)kvh * T_LEN * HEAD_DIM;
  const half_t* vbase = vT + (size_t)kvh * HEAD_DIM * T_LEN;

  auto process = [&](int kb) {
    // ---- S = Q * K^T over 32 keys (two 16-key N blocks) ----
    v8f Sa = zero_v8f(), Sb = zero_v8f();
#pragma unroll
    for (int c = 0; c < 4; ++c) {
      v16h kfA = *(const v16h*)(kbase + (size_t)(kb + n) * HEAD_DIM + 32 * c + koff);
      v16h kfB = *(const v16h*)(kbase + (size_t)(kb + 16 + n) * HEAD_DIM + 32 * c + koff);
      Sa = __builtin_amdgcn_wmma_f32_16x16x32_f16(false, qa[c], false, kfA,
                                                  (short)0, Sa, false, false);
      Sb = __builtin_amdgcn_wmma_f32_16x16x32_f16(false, qa[c], false, kfB,
                                                  (short)0, Sb, false, false);
    }

    __syncthreads();  // plds reuse guard (single wave: cheap)

    // ---- mask + online softmax; rows live across lanes (C-layout) ----
#pragma unroll
    for (int r = 0; r < 8; ++r) {
      const int qi = q0 + mbase + r;
      const int k1 = kb + n;
      const int k2 = kb + 16 + n;
      const bool a1 = (k1 <= qi) && ((k1 >= qi - (WINDOW - 1)) || (k1 < NUM_META));
      const bool a2 = (k2 <= qi) && ((k2 >= qi - (WINDOW - 1)) || (k2 < NUM_META));
      const float sa = a1 ? Sa[r] : -1e30f;   // select, no add
      const float sb = a2 ? Sb[r] : -1e30f;

      const float rm    = lane16_max(fmaxf(sa, sb));
      const float mnew  = fmaxf(m_run[r], rm);
      const float alpha = fast_exp2(m_run[r] - mnew);  // base-2 domain
      m_run[r] = mnew;

      plds[(mbase + r) * 32 + n]      = (half_t)fast_exp2(sa - mnew);
      plds[(mbase + r) * 32 + 16 + n] = (half_t)fast_exp2(sb - mnew);

      accl[r] *= alpha;
#pragma unroll
      for (int nc = 0; nc < 8; ++nc) acc[nc][r] *= alpha;
    }

    __syncthreads();  // all P stores visible before A-fragment reload

    // ---- P A-fragment via LDS transpose ----
    const half_t* prow = plds + n * 32;
    v8h p0 = *(const v8h*)(prow + koff);
    v8h p1 = *(const v8h*)(prow + koff + 8);
    v16h pf = cat16(p0, p1);

    // rowsum:  accl += P @ ones
    accl = __builtin_amdgcn_wmma_f32_16x16x32_f16(false, pf, false, onesf,
                                                  (short)0, accl, false, false);

    // acc += P @ V
#pragma unroll
    for (int nc = 0; nc < 8; ++nc) {
      v16h vf = *(const v16h*)(vbase + (size_t)(nc * 16 + n) * T_LEN + kb + koff);
      acc[nc] = __builtin_amdgcn_wmma_f32_16x16x32_f16(false, pf, false, vf,
                                                       (short)0, acc[nc], false, false);
    }
  };

  // ---- block schedule: meta blocks first, then window-aligned range ----
  const int qmax = q0 + 15;
  int wlo = q0 - (WINDOW - 1);
  if (wlo < 0) wlo = 0;
  const int kb_w = (wlo >> 5) << 5;

  for (int kb = 0; kb < kb_w && kb < NUM_META; kb += 32) process(kb);
  for (int kb = kb_w; kb <= qmax; kb += 32) process(kb);

  // ---- normalize and write out: [T][H*D] fp32 ----
#pragma unroll
  for (int r = 0; r < 8; ++r) {
    const float inv = 1.0f / accl[r];
    const size_t row = (size_t)(q0 + mbase + r) * (NUM_HEADS * HEAD_DIM) +
                       (size_t)h * HEAD_DIM;
#pragma unroll
    for (int nc = 0; nc < 8; ++nc)
      out[row + nc * 16 + n] = acc[nc][r] * inv;
  }
}

// ---------------------------------------------------------------------------
extern "C" void kernel_launch(void* const* d_in, const int* in_sizes, int n_in,
                              void* d_out, int out_size, void* d_ws, size_t ws_size,
                              hipStream_t stream) {
  const float* q   = (const float*)d_in[0];
  const float* k   = (const float*)d_in[1];
  const float* v   = (const float*)d_in[2];
  const float* qw  = (const float*)d_in[3];
  const float* kw  = (const float*)d_in[4];
  const int*   pos = (const int*)d_in[5];
  float* out = (float*)d_out;

  char* ws = (char*)d_ws;
  const size_t q_bytes = (size_t)NUM_HEADS * T_LEN * HEAD_DIM * sizeof(half_t);
  const size_t k_bytes = (size_t)NUM_KVH   * T_LEN * HEAD_DIM * sizeof(half_t);
  half_t* qh = (half_t*)ws;
  half_t* kh = (half_t*)(ws + q_bytes);
  half_t* vT = (half_t*)(ws + q_bytes + k_bytes);

  hymba_prep_kernel<<<dim3(T_LEN, NUM_HEADS + 2 * NUM_KVH), 128, 0, stream>>>(
      q, k, v, qw, kw, pos, qh, kh, vT);

  hymba_attn_kernel<<<dim3(T_LEN / 16, NUM_HEADS), 32, 0, stream>>>(
      qh, kh, vT, out);
}
